// Transformer_3118146257601
// MI455X (gfx1250) — matrix operand order repaired and checked by reading, not compiled, not run on hardware
//
#include <hip/hip_runtime.h>
#include <hip/hip_bf16.h>
#include <math.h>

#define D       256
#define NH      8
#define HD      32
#define DFF     2048
#define S_LEN   1024
#define BATCH   8
#define TQ      100
#define LENC    6
#define LDEC    6
#define BD      (BATCH * D)   // token row stride in [t, B, D] layout = 2048 elems
#define DEC_VS  128           // padded key stride for decoder self-attn V^T

typedef __attribute__((ext_vector_type(16))) __bf16 v16bf;
typedef __attribute__((ext_vector_type(8)))  __bf16 v8bf;
typedef __attribute__((ext_vector_type(2)))  __bf16 v2bf;
typedef __attribute__((ext_vector_type(8)))  float  v8f;

// ---------------------------------------------------------------------------
// fp32 -> bf16 conversions
// ---------------------------------------------------------------------------
static __device__ __forceinline__ __bf16 f2bf(float f) {
  unsigned u = __builtin_bit_cast(unsigned, f);
  unsigned r = u + 0x7FFFu + ((u >> 16) & 1u);
  unsigned short s = (unsigned short)(r >> 16);
  return __builtin_bit_cast(__bf16, s);
}

// packed pair conversion: prefer v_cvt_pk_bf16_f32 when the builtin exists
static __device__ __forceinline__ v2bf pk2bf(float x, float y) {
#if __has_builtin(__builtin_amdgcn_cvt_pk_bf16_f32)
  auto t = __builtin_amdgcn_cvt_pk_bf16_f32(x, y);
  static_assert(sizeof(t) == 4, "cvt_pk_bf16 size");
  return __builtin_bit_cast(v2bf, t);
#else
  v2bf r; r[0] = f2bf(x); r[1] = f2bf(y); return r;
#endif
}

// DPP lane shuffle (VALU path; no LDS, no dscnt waits)
template <int CTRL>
static __device__ __forceinline__ float dppf(float x) {
  return __builtin_bit_cast(float,
    __builtin_amdgcn_update_dpp(0, __builtin_bit_cast(int, x), CTRL, 0xF, 0xF, true));
}
// reduce across each 16-lane half: xor1, xor2, cross-quad, cross-octet
static __device__ __forceinline__ float red_max16(float x) {
  x = fmaxf(x, dppf<0xB1>(x));   // quad_perm [1,0,3,2]
  x = fmaxf(x, dppf<0x4E>(x));   // quad_perm [2,3,0,1]
  x = fmaxf(x, dppf<0x141>(x));  // row_half_mirror
  x = fmaxf(x, dppf<0x140>(x));  // row_mirror
  return x;
}
static __device__ __forceinline__ float red_sum16(float x) {
  x += dppf<0xB1>(x);
  x += dppf<0x4E>(x);
  x += dppf<0x141>(x);
  x += dppf<0x140>(x);
  return x;
}

union AFrag   { v16bf v; v2bf p[8]; };
union BF16x16 { v16bf v; v8bf h2[2]; v2bf p[8]; };

// ---------------------------------------------------------------------------
// Weight pre-conversion: fp32 -> bf16, 8 elements / thread, vector I/O
// ---------------------------------------------------------------------------
__global__ void cvt_w(const float* __restrict__ in, __bf16* __restrict__ out, int n8)
{
  int i = blockIdx.x * blockDim.x + threadIdx.x;
  if (i < n8) {
    const float4* p = (const float4*)in + 2 * (size_t)i;
    float4 a = p[0], b = p[1];
    union { v8bf v; v2bf p2[4]; } o;
    o.p2[0] = pk2bf(a.x, a.y); o.p2[1] = pk2bf(a.z, a.w);
    o.p2[2] = pk2bf(b.x, b.y); o.p2[3] = pk2bf(b.z, b.w);
    ((v8bf*)out)[i] = o.v;
  }
}

// ---------------------------------------------------------------------------
// Generic linear: C[M,N] = act((A[M,K] @ W[N,K]^T + bias[N]) * alpha)
// A is fp32 (converted on the fly, amortized over 4 WMMAs); W is pre-converted
// bf16, so B-fragments are pure vector loads. 4 waves / block; each wave owns
// a 16x64 tile -> 4 v_wmma_f32_16x16x32_bf16 per 32-wide k-step.
// Fragment element mapping used by this kernel:
//   A (16x32, bf16): lane holds row m=lane&15; k(j) = j + 16*(lane>>4)
//   B (32x16, bf16): lane holds col n=lane&15; k(j) = j + 16*(lane>>4)
//   C (16x16, f32) : element r -> row m = r + 8*(lane>>4), col n = lane&15
// MODE: 0 = fp32 row-major out, 1 = bf16 row-major out (Q/K operands),
//       2 = bf16 transposed-V out: C[(b*D + n) * vs + s], token m = s*B + b
// ---------------------------------------------------------------------------
template <int MODE>
__global__ void linear_wmma(const float* __restrict__ A, const __bf16* __restrict__ W,
                            const float* __restrict__ bias, void* __restrict__ Cout,
                            int M, int N, int K, int relu, float alpha, int vs)
{
  const int lane = threadIdx.x & 31;
  const int wave = threadIdx.x >> 5;
  const int lo = lane & 15, hi = lane >> 4;
  const int m0 = blockIdx.y * 64 + wave * 16;
  const int n0 = blockIdx.x * 64;
  if (m0 >= M) return;                       // wave-uniform: EXEC stays all-ones

  int ra = m0 + lo; if (ra > M - 1) ra = M - 1;
  const float* arow = A + (size_t)ra * K;
  const __bf16* wrow0 = W + (size_t)(n0 + lo) * K;
  const __bf16* wrow1 = wrow0 + (size_t)16 * K;
  const __bf16* wrow2 = wrow0 + (size_t)32 * K;
  const __bf16* wrow3 = wrow0 + (size_t)48 * K;

  v8f c0 = {0.f,0.f,0.f,0.f,0.f,0.f,0.f,0.f};
  v8f c1 = c0, c2 = c0, c3 = c0;

  for (int k0 = 0; k0 < K; k0 += 32) {
    // A fragment: two 8-float runs at k0+16*hi and k0+16*hi+8 (16B-aligned)
    const float4* ar1 = (const float4*)(arow + k0 + 16 * hi);
    const float4* ar2 = (const float4*)(arow + k0 + 16 * hi + 8);
    float4 fa0 = ar1[0], fa1 = ar1[1], fa2 = ar2[0], fa3 = ar2[1];
    AFrag a;
    a.p[0] = pk2bf(fa0.x, fa0.y); a.p[1] = pk2bf(fa0.z, fa0.w);
    a.p[2] = pk2bf(fa1.x, fa1.y); a.p[3] = pk2bf(fa1.z, fa1.w);
    a.p[4] = pk2bf(fa2.x, fa2.y); a.p[5] = pk2bf(fa2.z, fa2.w);
    a.p[6] = pk2bf(fa3.x, fa3.y); a.p[7] = pk2bf(fa3.z, fa3.w);

    const __bf16* wr[4] = {wrow0, wrow1, wrow2, wrow3};
    v8f* cc[4] = {&c0, &c1, &c2, &c3};
#pragma unroll
    for (int t = 0; t < 4; ++t) {
      // B fragment: 16 consecutive bf16 at k0+16*hi (32B-aligned) — no converts
      const __bf16* br = wr[t] + k0 + 16 * hi;
      BF16x16 b;
      b.h2[0] = *(const v8bf*)(br);
      b.h2[1] = *(const v8bf*)(br + 8);
      *cc[t] = __builtin_amdgcn_wmma_f32_16x16x32_bf16(false, a.v, false, b.v,
                                                       (short)0, *cc[t], false, false);
    }
  }

  const v8f* cc[4] = {&c0, &c1, &c2, &c3};
  const float bv[4] = {bias[n0 + lo], bias[n0 + 16 + lo],
                       bias[n0 + 32 + lo], bias[n0 + 48 + lo]};
  const bool full = (m0 + 16 <= M);

#pragma unroll
  for (int r = 0; r < 8; ++r) {
    int m = m0 + r + 8 * hi;
    if (full || m < M) {
#pragma unroll
      for (int t = 0; t < 4; ++t) {
        float v = ((*cc[t])[r] + bv[t]) * alpha;
        if (relu) v = fmaxf(v, 0.f);
        int col = n0 + 16 * t + lo;
        if (MODE == 0) {
          ((float*)Cout)[(size_t)m * N + col] = v;
        } else if (MODE == 1) {
          ((__bf16*)Cout)[(size_t)m * N + col] = f2bf(v);
        } else {
          // token m = s*BATCH + b  ->  V^T [(b*D + col) * vs + s]
          ((__bf16*)Cout)[(size_t)((m & (BATCH - 1)) * D + col) * vs + (m >> 3)] = f2bf(v);
        }
      }
    }
  }
}

// ---------------------------------------------------------------------------
// Wave-level flash attention over bf16 Q/K (row-major, [t,B,D]) and V^T
// ([(b*D+dim)*vs + key]). Scale pre-folded into Q. One wave per 16-query tile
// per (batch, head); streams keys in 64-wide chunks: 4 score WMMAs + DPP-based
// online softmax + 4 P.V WMMAs. Scores never materialized.
// ---------------------------------------------------------------------------
__global__ void flash_attn(const __bf16* __restrict__ Q, const __bf16* __restrict__ Km,
                           const __bf16* __restrict__ Vt, float* __restrict__ O,
                           int Tq, int S, int vs)
{
  __shared__ __attribute__((aligned(16))) __bf16 P[16 * 64];
  const int lane = threadIdx.x & 31;
  const int lo = lane & 15, hi = lane >> 4;
  const int bh = blockIdx.y;
  const int b = bh / NH, h = bh % NH;
  const int coff = b * D + h * HD;
  const int q0 = blockIdx.x * 16;

  // Q fragment (A layout): two 8-element bf16 runs per lane (16B-aligned)
  int qm = q0 + lo; if (qm > Tq - 1) qm = Tq - 1;
  const __bf16* qrow = Q + (size_t)qm * BD + coff;
  BF16x16 aq;
  aq.h2[0] = *(const v8bf*)(qrow + 16 * hi);
  aq.h2[1] = *(const v8bf*)(qrow + 16 * hi + 8);

  float mrow[8], lrow[8];
  v8f o0 = {0.f,0.f,0.f,0.f,0.f,0.f,0.f,0.f};
  v8f o1 = o0;
#pragma unroll
  for (int r = 0; r < 8; ++r) { mrow[r] = -1e30f; lrow[r] = 0.f; }

  const __bf16* v0r = Vt + (size_t)(coff + lo) * vs;        // dims  0..15 (this lane: lo)
  const __bf16* v1r = Vt + (size_t)(coff + 16 + lo) * vs;   // dims 16..31

  for (int c = 0; c < S; c += 64) {
    // ---- scores: 4 x 16-key column blocks ----
    v8f cf[4];
#pragma unroll
    for (int t = 0; t < 4; ++t) {
      int key = c + 16 * t + lo;
      int kc = key < S ? key : S - 1;
      const __bf16* kr = Km + (size_t)kc * BD + coff + 16 * hi;
      BF16x16 kb;
      kb.h2[0] = *(const v8bf*)(kr);
      kb.h2[1] = *(const v8bf*)(kr + 8);
      v8f z = {0.f,0.f,0.f,0.f,0.f,0.f,0.f,0.f};
      cf[t] = __builtin_amdgcn_wmma_f32_16x16x32_bf16(false, aq.v, false, kb.v,
                                                      (short)0, z, false, false);
    }

    if (c + 64 > S) {            // uniform tail branch: mask invalid keys
#pragma unroll
      for (int t = 0; t < 4; ++t) {
        bool inv = (c + 16 * t + lo) >= S;
#pragma unroll
        for (int r = 0; r < 8; ++r) cf[t][r] = inv ? -1e30f : cf[t][r];
      }
    }

    // ---- online softmax per row (row lives across a 16-lane half) ----
#pragma unroll
    for (int r = 0; r < 8; ++r) {
      float vm = fmaxf(fmaxf(cf[0][r], cf[1][r]), fmaxf(cf[2][r], cf[3][r]));
      vm = red_max16(vm);
      float mnew = fmaxf(mrow[r], vm);
      float p0 = __expf(cf[0][r] - mnew);
      float p1 = __expf(cf[1][r] - mnew);
      float p2 = __expf(cf[2][r] - mnew);
      float p3 = __expf(cf[3][r] - mnew);
      float ps = red_sum16((p0 + p1) + (p2 + p3));
      float corr = __expf(mrow[r] - mnew);
      lrow[r] = lrow[r] * corr + ps;
      mrow[r] = mnew;
      o0[r] = o0[r] * corr;
      o1[r] = o1[r] * corr;
      int prow = (r + 8 * hi) * 64;
      P[prow + lo]      = f2bf(p0);
      P[prow + 16 + lo] = f2bf(p1);
      P[prow + 32 + lo] = f2bf(p2);
      P[prow + 48 + lo] = f2bf(p3);
    }
    __syncthreads();

    // ---- O += P @ V : P C-frag -> two A-frags via LDS; V^T vector loads ----
    BF16x16 ap0, ap1, vb00, vb01, vb10, vb11;
    const __bf16* pr = P + lo * 64 + 16 * hi;
    ap0.h2[0] = *(const v8bf*)(pr);
    ap0.h2[1] = *(const v8bf*)(pr + 8);
    ap1.h2[0] = *(const v8bf*)(pr + 32);
    ap1.h2[1] = *(const v8bf*)(pr + 40);
    const __bf16* p0r = v0r + c + 16 * hi;
    const __bf16* p1r = v1r + c + 16 * hi;
    vb00.h2[0] = *(const v8bf*)(p0r);       vb00.h2[1] = *(const v8bf*)(p0r + 8);
    vb10.h2[0] = *(const v8bf*)(p0r + 32);  vb10.h2[1] = *(const v8bf*)(p0r + 40);
    vb01.h2[0] = *(const v8bf*)(p1r);       vb01.h2[1] = *(const v8bf*)(p1r + 8);
    vb11.h2[0] = *(const v8bf*)(p1r + 32);  vb11.h2[1] = *(const v8bf*)(p1r + 40);
    o0 = __builtin_amdgcn_wmma_f32_16x16x32_bf16(false, ap0.v, false, vb00.v, (short)0, o0, false, false);
    o0 = __builtin_amdgcn_wmma_f32_16x16x32_bf16(false, ap1.v, false, vb10.v, (short)0, o0, false, false);
    o1 = __builtin_amdgcn_wmma_f32_16x16x32_bf16(false, ap0.v, false, vb01.v, (short)0, o1, false, false);
    o1 = __builtin_amdgcn_wmma_f32_16x16x32_bf16(false, ap1.v, false, vb11.v, (short)0, o1, false, false);
    __syncthreads();
  }

#pragma unroll
  for (int r = 0; r < 8; ++r) {
    int q = q0 + r + 8 * hi;
    if (q < Tq) {
      float inv = 1.f / lrow[r];
      float* orow = O + (size_t)q * BD + coff;
      orow[lo]      = o0[r] * inv;
      orow[16 + lo] = o1[r] * inv;
    }
  }
}

// ---------------------------------------------------------------------------
// Elementwise helpers
// ---------------------------------------------------------------------------
__global__ void add_vec(const float* __restrict__ a, const float* __restrict__ b,
                        float* __restrict__ c, int n)
{
  int i = blockIdx.x * blockDim.x + threadIdx.x;
  if (i < n) c[i] = a[i] + b[i];
}

__global__ void fill_zero(float* __restrict__ p, int n)
{
  int i = blockIdx.x * blockDim.x + threadIdx.x;
  if (i < n) p[i] = 0.f;
}

// out = LayerNorm(x + r) * g + b ; r may be null. One block per token (D=256).
__global__ void ln_res(const float* __restrict__ x, const float* __restrict__ r,
                       const float* __restrict__ g, const float* __restrict__ bt,
                       float* __restrict__ out, int ntok)
{
  __shared__ float sh[D];
  int t = blockIdx.x, i = threadIdx.x;
  float v = x[(size_t)t * D + i] + (r ? r[(size_t)t * D + i] : 0.f);
  sh[i] = v; __syncthreads();
  for (int s = D / 2; s > 0; s >>= 1) { if (i < s) sh[i] += sh[i + s]; __syncthreads(); }
  float mean = sh[0] * (1.f / D); __syncthreads();
  float d = v - mean;
  sh[i] = d * d; __syncthreads();
  for (int s = D / 2; s > 0; s >>= 1) { if (i < s) sh[i] += sh[i + s]; __syncthreads(); }
  float var = sh[0] * (1.f / D);
  out[(size_t)t * D + i] = d * rsqrtf(var + 1e-5f) * g[i] + bt[i];
}

// ---------------------------------------------------------------------------
// Host orchestration
// ---------------------------------------------------------------------------
static inline void linear_f(const float* A, const __bf16* W, const float* bias, float* C,
                            int M, int N, int K, int relu, hipStream_t s)
{
  dim3 grid(N / 64, (M + 63) / 64);
  linear_wmma<0><<<grid, 128, 0, s>>>(A, W, bias, C, M, N, K, relu, 1.0f, 0);
}
static inline void linear_b(const float* A, const __bf16* W, const float* bias, __bf16* C,
                            int M, int N, int K, float alpha, hipStream_t s)
{
  dim3 grid(N / 64, (M + 63) / 64);
  linear_wmma<1><<<grid, 128, 0, s>>>(A, W, bias, C, M, N, K, 0, alpha, 0);
}
static inline void linear_vt(const float* A, const __bf16* W, const float* bias, __bf16* C,
                             int M, int N, int K, int vs, hipStream_t s)
{
  dim3 grid(N / 64, (M + 63) / 64);
  linear_wmma<2><<<grid, 128, 0, s>>>(A, W, bias, C, M, N, K, 0, 1.0f, vs);
}

extern "C" void kernel_launch(void* const* d_in, const int* in_sizes, int n_in,
                              void* d_out, int out_size, void* d_ws, size_t ws_size,
                              hipStream_t stream)
{
  (void)in_sizes; (void)n_in; (void)out_size; (void)ws_size;

  const float* src  = (const float*)d_in[0];
  const float* pos  = (const float*)d_in[1];
  const float* qenc = (const float*)d_in[2];

  // params in setup_inputs() insertion order
  int idx = 3;
  const float* enc[LENC][12]; // in_w,in_b,out_w,out_b,l1_w,l1_b,l2_w,l2_b,n1_g,n1_b,n2_g,n2_b
  for (int l = 0; l < LENC; ++l)
    for (int k = 0; k < 12; ++k) enc[l][k] = (const float*)d_in[idx++];
  const float* dec[LDEC][18]; // sa(4), ca(4), l1_w,l1_b,l2_w,l2_b, n1_g,n1_b,n2_g,n2_b,n3_g,n3_b
  for (int l = 0; l < LDEC; ++l)
    for (int k = 0; k < 18; ++k) dec[l][k] = (const float*)d_in[idx++];
  const float* dn_g = (const float*)d_in[idx++];
  const float* dn_b = (const float*)d_in[idx++];

  const size_t NE = (size_t)S_LEN * BATCH * D;  // 2,097,152 elems
  const size_t ND = (size_t)TQ * BATCH * D;     //   204,800 elems
  const size_t NVD = (size_t)BATCH * D * DEC_VS; // 262,144 elems (dec V^T, padded)
  float* ws = (float*)d_ws;
  float* x    = ws; ws += NE;                    // encoder activations / memory
  float* tq   = ws; ws += NE;                    // enc tmp / (mem + pos) in decoder phase
  float* attn = ws; ws += NE;
  float* hbuf = ws; ws += (size_t)2048 * DFF;    // FFN hidden chunk
  float* dx   = ws; ws += ND;
  float* dq   = ws; ws += ND;
  float* dt   = ws; ws += ND;
  float* datn = ws; ws += ND;
  __bf16* Qp  = (__bf16*)ws; ws += NE / 2;       // bf16 buffers (half-size in floats)
  __bf16* Kp  = (__bf16*)ws; ws += NE / 2;
  __bf16* Vtp = (__bf16*)ws; ws += NE / 2;       // V^T, stride S_LEN
  __bf16* dQ  = (__bf16*)ws; ws += ND / 2;
  __bf16* dK  = (__bf16*)ws; ws += ND / 2;
  __bf16* dVt = (__bf16*)ws; ws += NVD / 2;      // V^T, stride DEC_VS (padded)
  __bf16* wpool = (__bf16*)ws;                   // bf16 weight pool (converted once)

  const int NTOK_E = S_LEN * BATCH;  // 8192
  const int NTOK_D = TQ * BATCH;     // 800
  const float scale = 0.17677669529663687f;  // 1/sqrt(32)
  const int EBLK = 256;

  // ---- pre-convert all GEMM weights to bf16 (one pass each per launch) ----
  __bf16* wcur = wpool;
  auto cvt = [&](const float* w, size_t n) -> const __bf16* {
    __bf16* dst = wcur; wcur += n;
    int n8 = (int)(n / 8);
    cvt_w<<<(n8 + EBLK - 1) / EBLK, EBLK, 0, stream>>>(w, dst, n8);
    return dst;
  };
  const __bf16 *encw[LENC][4], *decw[LDEC][6];
  for (int l = 0; l < LENC; ++l) {
    encw[l][0] = cvt(enc[l][0], (size_t)3 * D * D);   // in_w (Q|K|V)
    encw[l][1] = cvt(enc[l][2], (size_t)D * D);       // out_w
    encw[l][2] = cvt(enc[l][4], (size_t)DFF * D);     // l1_w
    encw[l][3] = cvt(enc[l][6], (size_t)D * DFF);     // l2_w
  }
  for (int l = 0; l < LDEC; ++l) {
    decw[l][0] = cvt(dec[l][0], (size_t)3 * D * D);   // sa in_w
    decw[l][1] = cvt(dec[l][2], (size_t)D * D);       // sa out_w
    decw[l][2] = cvt(dec[l][4], (size_t)3 * D * D);   // ca in_w
    decw[l][3] = cvt(dec[l][6], (size_t)D * D);       // ca out_w
    decw[l][4] = cvt(dec[l][8], (size_t)DFF * D);     // l1_w
    decw[l][5] = cvt(dec[l][10], (size_t)D * DFF);    // l2_w
  }

  hipMemcpyAsync(x, src, NE * sizeof(float), hipMemcpyDeviceToDevice, stream);

  // ---------------- encoder ----------------
  for (int l = 0; l < LENC; ++l) {
    const float* const* p = enc[l];
    const __bf16* const* w = encw[l];
    add_vec<<<(NTOK_E * D + EBLK - 1) / EBLK, EBLK, 0, stream>>>(x, pos, tq, NTOK_E * D);
    linear_b (tq, w[0],             p[1],         Qp,  NTOK_E, D, D, scale, stream);
    linear_b (tq, w[0] + D * D,     p[1] + D,     Kp,  NTOK_E, D, D, 1.0f, stream);
    linear_vt(x,  w[0] + 2 * D * D, p[1] + 2 * D, Vtp, NTOK_E, D, D, S_LEN, stream);
    flash_attn<<<dim3(S_LEN / 16, BATCH * NH), 32, 0, stream>>>(Qp, Kp, Vtp, attn,
                                                                S_LEN, S_LEN, S_LEN);
    linear_f(attn, w[1], p[3], tq, NTOK_E, D, D, 0, stream);
    ln_res<<<NTOK_E, D, 0, stream>>>(x, tq, p[8], p[9], x, NTOK_E);
    for (int c = 0; c < 4; ++c) {  // FFN in 2048-row chunks to bound workspace
      size_t m0 = (size_t)c * 2048;
      linear_f(x + m0 * D, w[2], p[5], hbuf, 2048, DFF, D, 1, stream);
      linear_f(hbuf, w[3], p[7], tq + m0 * D, 2048, D, DFF, 0, stream);
    }
    ln_res<<<NTOK_E, D, 0, stream>>>(x, tq, p[10], p[11], x, NTOK_E);
  }

  // ---------------- decoder ----------------
  add_vec<<<(NTOK_E * D + EBLK - 1) / EBLK, EBLK, 0, stream>>>(x, pos, tq, NTOK_E * D); // mem+pos
  fill_zero<<<(NTOK_D * D + EBLK - 1) / EBLK, EBLK, 0, stream>>>(dx, NTOK_D * D);
  fill_zero<<<((int)(NVD / 2) + EBLK - 1) / EBLK, EBLK, 0, stream>>>((float*)dVt, (int)(NVD / 2));

  for (int l = 0; l < LDEC; ++l) {
    const float* const* p = dec[l];
    const __bf16* const* w = decw[l];
    // self-attention: q = x + qenc, v = x
    add_vec<<<(NTOK_D * D + EBLK - 1) / EBLK, EBLK, 0, stream>>>(dx, qenc, dq, NTOK_D * D);
    linear_b (dq, w[0],             p[1],         dQ,  NTOK_D, D, D, scale, stream);
    linear_b (dq, w[0] + D * D,     p[1] + D,     dK,  NTOK_D, D, D, 1.0f, stream);
    linear_vt(dx, w[0] + 2 * D * D, p[1] + 2 * D, dVt, NTOK_D, D, D, DEC_VS, stream);
    flash_attn<<<dim3((TQ + 15) / 16, BATCH * NH), 32, 0, stream>>>(dQ, dK, dVt, datn,
                                                                    TQ, TQ, DEC_VS);
    linear_f(datn, w[1], p[3], dt, NTOK_D, D, D, 0, stream);
    ln_res<<<NTOK_D, D, 0, stream>>>(dx, dt, p[12], p[13], dx, NTOK_D);

    // cross-attention: q = x + qenc, k = mem + pos, v = mem
    add_vec<<<(NTOK_D * D + EBLK - 1) / EBLK, EBLK, 0, stream>>>(dx, qenc, dq, NTOK_D * D);
    linear_b (dq, w[2],             p[5],         dQ,  NTOK_D, D, D, scale, stream);
    linear_b (tq, w[2] + D * D,     p[5] + D,     Kp,  NTOK_E, D, D, 1.0f, stream);
    linear_vt(x,  w[2] + 2 * D * D, p[5] + 2 * D, Vtp, NTOK_E, D, D, S_LEN, stream);
    flash_attn<<<dim3((TQ + 15) / 16, BATCH * NH), 32, 0, stream>>>(dQ, Kp, Vtp, datn,
                                                                    TQ, S_LEN, S_LEN);
    linear_f(datn, w[3], p[7], dt, NTOK_D, D, D, 0, stream);
    ln_res<<<NTOK_D, D, 0, stream>>>(dx, dt, p[14], p[15], dx, NTOK_D);

    // FFN
    linear_f(dx, w[4], p[9], hbuf, NTOK_D, DFF, D, 1, stream);
    linear_f(hbuf, w[5], p[11], dt, NTOK_D, D, DFF, 0, stream);
    ln_res<<<NTOK_D, D, 0, stream>>>(dx, dt, p[16], p[17], dx, NTOK_D);

    // intermediate output: final norm of current decoder state
    ln_res<<<NTOK_D, D, 0, stream>>>(dx, nullptr, dn_g, dn_b,
                                     (float*)d_out + (size_t)l * NTOK_D * D, NTOK_D);
  }
}
